// RadixMLPQwen3Attention_39367670235187
// MI455X (gfx1250) — compile-verified, matrix operand order checked by reading, not run on hardware
//
#include <hip/hip_runtime.h>
#include <hip/hip_bf16.h>

typedef __attribute__((ext_vector_type(16))) __bf16 v16bf;
typedef __attribute__((ext_vector_type(8)))  float  v8f;

// ---- problem dims (fixed by reference) ----
#define NH    12      // H
#define NKV   2       // KV heads
#define DH    128     // head dim
#define HID   1536
#define BB    4
#define SS    2048
#define TT    (BB*SS)     // 8192
#define HD    (NH*DH)     // 1536
#define KVD   (NKV*DH)    // 256
#define GQA   (NH/NKV)    // 6

// padded LDS row strides (elements) to spread banks
#define KROWP 136         // 32 K-rows of 128 bf16, padded (272B stride)
#define VROWP 40          // 128 V-rows of 32 bf16, padded (80B stride)

// ============================================================
// bf16 fragment loader following documented 16-bit A-matrix
// layout: lane<16 -> K = {0..7, 16..23}; lane>=16 -> K = {8..15, 24..31}
// VGPR j packs (K, K+1).  Coalesces to 2x b128 loads (global or ds).
// ============================================================
__device__ inline v16bf load_frag(const __bf16* __restrict__ rowptr, int half) {
    v16bf f;
#pragma unroll
    for (int j = 0; j < 4; ++j) {
        int kk = 2*j + 8*half;
        f[2*j]     = rowptr[kk];
        f[2*j + 1] = rowptr[kk + 1];
        f[8 + 2*j]     = rowptr[16 + kk];
        f[8 + 2*j + 1] = rowptr[16 + kk + 1];
    }
    return f;
}

// ============================================================
// Stage 1 kernels: precision conversion / weight transposes
// ============================================================
__global__ void k_f32_to_bf16(const float* __restrict__ src, __bf16* __restrict__ dst, int n) {
    int i = blockIdx.x * blockDim.x + threadIdx.x;
    if (i < n) dst[i] = (__bf16)src[i];
}

// dst[n*K + k] = src[k*N + n]   (W is K x N row-major; dst is N x K)
__global__ void k_transpose_bf16(const float* __restrict__ src, __bf16* __restrict__ dst, int K, int N) {
    int i = blockIdx.x * blockDim.x + threadIdx.x;
    if (i >= N * K) return;
    int n = i / K, k = i % K;
    dst[i] = (__bf16)src[(size_t)k * N + n];
}

// ============================================================
// Stage 2/5: WMMA GEMM  C[MxN] = A[MxK] * Bt[NxK]^T  (bf16 in, f32 out)
// Register-blocked: each wave produces a 32x64 tile (2x4 WMMA tiles).
// Per 32-wide K chunk: 6 fragment loads feed 8 v_wmma ops.
// ============================================================
__global__ __launch_bounds__(256) void k_gemm_wmma(const __bf16* __restrict__ A,
                                                   const __bf16* __restrict__ Bt,
                                                   float* __restrict__ C,
                                                   int M, int N, int K) {
    int wave = blockIdx.x * (blockDim.x >> 5) + (threadIdx.x >> 5);
    int lane = threadIdx.x & 31;
    int half = lane >> 4, lrow = lane & 15;
    int tiles_n = N >> 6;                         // 64-col macro tiles
    int tiles_total = (M >> 5) * tiles_n;         // 32-row macro tiles
    if (wave >= tiles_total) return;              // wave-uniform
    int tm = wave / tiles_n, tn = wave % tiles_n;

    const __bf16* arow0 = A  + (size_t)(tm * 32 +      lrow) * K;
    const __bf16* arow1 = A  + (size_t)(tm * 32 + 16 + lrow) * K;
    const __bf16* brow0 = Bt + (size_t)(tn * 64 +      lrow) * K;
    const __bf16* brow1 = Bt + (size_t)(tn * 64 + 16 + lrow) * K;
    const __bf16* brow2 = Bt + (size_t)(tn * 64 + 32 + lrow) * K;
    const __bf16* brow3 = Bt + (size_t)(tn * 64 + 48 + lrow) * K;

    v8f acc[2][4];
#pragma unroll
    for (int g = 0; g < 2; ++g)
#pragma unroll
        for (int u = 0; u < 4; ++u) { v8f z = {}; acc[g][u] = z; }

    for (int k0 = 0; k0 < K; k0 += 32) {
        // prefetch next chunk (gfx1250 global_prefetch path)
        __builtin_prefetch(arow0 + k0 + 64, 0, 1);
        __builtin_prefetch(brow0 + k0 + 64, 0, 1);

        v16bf a0 = load_frag(arow0 + k0, half);
        v16bf a1 = load_frag(arow1 + k0, half);

        v16bf b0 = load_frag(brow0 + k0, half);
        acc[0][0] = __builtin_amdgcn_wmma_f32_16x16x32_bf16(false, a0, false, b0, (short)0, acc[0][0], false, false);
        acc[1][0] = __builtin_amdgcn_wmma_f32_16x16x32_bf16(false, a1, false, b0, (short)0, acc[1][0], false, false);

        v16bf b1 = load_frag(brow1 + k0, half);
        acc[0][1] = __builtin_amdgcn_wmma_f32_16x16x32_bf16(false, a0, false, b1, (short)0, acc[0][1], false, false);
        acc[1][1] = __builtin_amdgcn_wmma_f32_16x16x32_bf16(false, a1, false, b1, (short)0, acc[1][1], false, false);

        v16bf b2 = load_frag(brow2 + k0, half);
        acc[0][2] = __builtin_amdgcn_wmma_f32_16x16x32_bf16(false, a0, false, b2, (short)0, acc[0][2], false, false);
        acc[1][2] = __builtin_amdgcn_wmma_f32_16x16x32_bf16(false, a1, false, b2, (short)0, acc[1][2], false, false);

        v16bf b3 = load_frag(brow3 + k0, half);
        acc[0][3] = __builtin_amdgcn_wmma_f32_16x16x32_bf16(false, a0, false, b3, (short)0, acc[0][3], false, false);
        acc[1][3] = __builtin_amdgcn_wmma_f32_16x16x32_bf16(false, a1, false, b3, (short)0, acc[1][3], false, false);
    }

#pragma unroll
    for (int g = 0; g < 2; ++g)
#pragma unroll
        for (int u = 0; u < 4; ++u)
#pragma unroll
            for (int i = 0; i < 8; ++i)
                C[(size_t)(tm * 32 + g * 16 + i + 8 * half) * N + tn * 64 + u * 16 + lrow]
                    = acc[g][u][i];
}

// ============================================================
// Stage 3a: fused RMSNorm + RoPE, one wave per (token, head) row of 128.
// ============================================================
__global__ __launch_bounds__(256) void k_rmsnorm_rope(const float* __restrict__ X,
                                                      const float* __restrict__ w,
                                                      const float* __restrict__ cosb,
                                                      const float* __restrict__ sinb,
                                                      __bf16* __restrict__ Y,
                                                      int nheads) {
    int wid  = threadIdx.x >> 5;
    int lane = threadIdx.x & 31;
    long row = (long)blockIdx.x * (blockDim.x >> 5) + wid;
    if (row >= (long)TT * nheads) return;
    int t = (int)(row / nheads);

    const float* x = X + row * DH;
    int dbase = lane * 4;
    float4 v = *(const float4*)(x + dbase);

    float ss = v.x*v.x + v.y*v.y + v.z*v.z + v.w*v.w;
#pragma unroll
    for (int m = 1; m < 32; m <<= 1) ss += __shfl_xor(ss, m, 32);
    float r = rsqrtf(ss * (1.0f / (float)DH) + 1e-6f);

    float n0 = v.x * r * w[dbase + 0];
    float n1 = v.y * r * w[dbase + 1];
    float n2 = v.z * r * w[dbase + 2];
    float n3 = v.w * r * w[dbase + 3];

    // exchange with rotate-half partner (lane ^ 16)
    float o0 = __shfl_xor(n0, 16, 32);
    float o1 = __shfl_xor(n1, 16, 32);
    float o2 = __shfl_xor(n2, 16, 32);
    float o3 = __shfl_xor(n3, 16, 32);

    int hd = dbase & 63;
    const float* cp = cosb + (size_t)t * 64 + hd;
    const float* sp = sinb + (size_t)t * 64 + hd;
    float c0 = cp[0], c1 = cp[1], c2 = cp[2], c3 = cp[3];
    float s0 = sp[0], s1 = sp[1], s2 = sp[2], s3 = sp[3];

    float r0, r1, r2, r3;
    if (lane < 16) {       // first half: x1*c - x2*s
        r0 = n0 * c0 - o0 * s0;  r1 = n1 * c1 - o1 * s1;
        r2 = n2 * c2 - o2 * s2;  r3 = n3 * c3 - o3 * s3;
    } else {               // second half: x1*s + x2*c  (x1 = partner)
        r0 = o0 * s0 + n0 * c0;  r1 = o1 * s1 + n1 * c1;
        r2 = o2 * s2 + n2 * c2;  r3 = o3 * s3 + n3 * c3;
    }
    __bf16* y = Y + row * DH + dbase;
    y[0] = (__bf16)r0; y[1] = (__bf16)r1; y[2] = (__bf16)r2; y[3] = (__bf16)r3;
}

// ============================================================
// Stage 3b: V transpose to [ (b*KV+kv)*D + d ][ s ]  bf16
// ============================================================
__global__ void k_v_transpose(const float* __restrict__ V, __bf16* __restrict__ Vt, int n) {
    int i = blockIdx.x * blockDim.x + threadIdx.x;
    if (i >= n) return;
    int t  = i / KVD;
    int rm = i % KVD;        // kv*D + d
    int b  = t / SS, s = t % SS;
    Vt[((size_t)b * KVD + rm) * SS + s] = (__bf16)V[i];
}

// ============================================================
// Stage 4: flash attention, one wave per (b,h, 16-query tile).
// Double-buffered async-to-LDS staging of K/V chunks:
//   while computing chunk i from LDS, chunk i+1 streams in via
//   global_load_async_to_lds_b128; s_wait_asynccnt 32 separates them
//   (in-order async completion: <=32 outstanding => current chunk done).
// 16 v_wmma per 32-key chunk (8 score + 8 PV), online softmax in-register.
// ============================================================
__global__ __launch_bounds__(32) void k_flash_attn(const __bf16* __restrict__ qh,
                                                   const __bf16* __restrict__ kh,
                                                   const __bf16* __restrict__ vt,
                                                   __bf16* __restrict__ aout) {
    __shared__ __bf16 sK[2][32 * KROWP];    // K chunk: 32 rows x 128, padded
    __shared__ __bf16 sV[2][128 * VROWP];   // V chunk: 128 d-rows x 32 keys, padded
    __shared__ float  sP[16 * 32];

    int lane = threadIdx.x;
    int half = lane >> 4, lrow = lane & 15;
    int qt = blockIdx.x;
    int bh = blockIdx.y;
    int b = bh / NH, h = bh % NH, kv = h / GQA;
    int t0 = b * SS + qt * 16;

    const __bf16* vbase = vt + (size_t)(b * NKV + kv) * DH * SS;

    // issue async LDS staging for the chunk starting at key s0n
    auto issue_chunk = [&](int s0n, int bufn) {
        // K: lane stages its own row (s0n+lane), 16 x b128
        const __bf16* gk = kh + (size_t)(b * SS + s0n + lane) * KVD + kv * DH;
        unsigned int lk = (unsigned int)(unsigned long long)&sK[bufn][lane * KROWP];
#pragma unroll
        for (int j = 0; j < 16; ++j) {
            unsigned long long ga = (unsigned long long)(const void*)(gk + j * 8);
            unsigned int la = lk + j * 16;
            asm volatile("global_load_async_to_lds_b128 %0, %1, off"
                         :: "v"(la), "v"(ga) : "memory");
        }
        // V: lane stages d-rows {4*lane .. 4*lane+3}, 4 x b128 each
#pragma unroll
        for (int jj = 0; jj < 4; ++jj) {
            const __bf16* gv = vbase + (size_t)(lane * 4 + jj) * SS + s0n;
            unsigned int lv = (unsigned int)(unsigned long long)&sV[bufn][(lane * 4 + jj) * VROWP];
#pragma unroll
            for (int p = 0; p < 4; ++p) {
                unsigned long long ga = (unsigned long long)(const void*)(gv + p * 8);
                unsigned int la = lv + p * 16;
                asm volatile("global_load_async_to_lds_b128 %0, %1, off"
                             :: "v"(la), "v"(ga) : "memory");
            }
        }
    };

    // Q fragments, 4 K-chunks of 32 over D=128
    const __bf16* qrow = qh + (size_t)(t0 + lrow) * HD + h * DH;
    v16bf qf[4];
#pragma unroll
    for (int c = 0; c < 4; ++c) qf[c] = load_frag(qrow + 32 * c, half);

    float mrow[8], lsum[8];
    v8f O[8];
#pragma unroll
    for (int c = 0; c < 8; ++c) { v8f z = {}; O[c] = z; }
#pragma unroll
    for (int i = 0; i < 8; ++i) { mrow[i] = -3.0e38f; lsum[i] = 0.f; }

    const float scale = 0.088388347648318447f;   // 1/sqrt(128)
    int kend = qt * 16 + 16;

    issue_chunk(0, 0);      // prime the pipeline

    for (int s0 = 0; s0 < kend; s0 += 32) {
        int buf = (s0 >> 5) & 1;
        if (s0 + 32 < kend) {
            issue_chunk(s0 + 32, buf ^ 1);                   // overlap next chunk
            asm volatile("s_wait_asynccnt 32" ::: "memory"); // current chunk landed
        } else {
            asm volatile("s_wait_asynccnt 0" ::: "memory");
        }

        // ---- scores: two 16x16 sub-tiles (Q from reg, K from LDS) ----
        v8f sc[2];
#pragma unroll
        for (int sub = 0; sub < 2; ++sub) {
            v8f s = {};
            const __bf16* krow = &sK[buf][(sub * 16 + lrow) * KROWP];
#pragma unroll
            for (int c = 0; c < 4; ++c) {
                v16bf kf = load_frag(krow + 32 * c, half);
                s = __builtin_amdgcn_wmma_f32_16x16x32_bf16(false, qf[c], false, kf,
                                                            (short)0, s, false, false);
            }
            sc[sub] = s;
        }
        // ---- scale + causal mask + chunk row-max ----
        float cmax[8];
#pragma unroll
        for (int i = 0; i < 8; ++i) {
            int qp = qt * 16 + i + 8 * half;
            float x0 = sc[0][i] * scale; if (s0 + lrow      > qp) x0 = -3.0e38f;
            float x1 = sc[1][i] * scale; if (s0 + 16 + lrow > qp) x1 = -3.0e38f;
            sc[0][i] = x0; sc[1][i] = x1;
            float m = fmaxf(x0, x1);
            m = fmaxf(m, __shfl_xor(m, 1, 32));
            m = fmaxf(m, __shfl_xor(m, 2, 32));
            m = fmaxf(m, __shfl_xor(m, 4, 32));
            m = fmaxf(m, __shfl_xor(m, 8, 32));
            cmax[i] = m;
        }
        // ---- online softmax update + stage P in LDS ----
#pragma unroll
        for (int i = 0; i < 8; ++i) {
            float mnew  = fmaxf(mrow[i], cmax[i]);
            float alpha = __expf(mrow[i] - mnew);
            float p0 = __expf(sc[0][i] - mnew);
            float p1 = __expf(sc[1][i] - mnew);
            float ps = p0 + p1;
            ps += __shfl_xor(ps, 1, 32);
            ps += __shfl_xor(ps, 2, 32);
            ps += __shfl_xor(ps, 4, 32);
            ps += __shfl_xor(ps, 8, 32);
            lsum[i] = lsum[i] * alpha + ps;
            mrow[i] = mnew;
#pragma unroll
            for (int c = 0; c < 8; ++c) O[c][i] *= alpha;
            sP[(i + 8 * half) * 32 + lrow]      = p0;
            sP[(i + 8 * half) * 32 + 16 + lrow] = p1;
        }
        __syncthreads();
        // ---- rebuild P as 16x32 A-fragment ----
        v16bf pf;
        const float* prow = sP + lrow * 32;
#pragma unroll
        for (int j = 0; j < 4; ++j) {
            int kk = 2 * j + 8 * half;
            pf[2*j]       = (__bf16)prow[kk];
            pf[2*j + 1]   = (__bf16)prow[kk + 1];
            pf[8 + 2*j]   = (__bf16)prow[16 + kk];
            pf[8 + 2*j+1] = (__bf16)prow[16 + kk + 1];
        }
        __syncthreads();
        // ---- O += P * V  (8 d-tiles of 16, V from LDS) ----
#pragma unroll
        for (int c = 0; c < 8; ++c) {
            const __bf16* vrow = &sV[buf][(c * 16 + lrow) * VROWP];
            v16bf vf = load_frag(vrow, half);
            O[c] = __builtin_amdgcn_wmma_f32_16x16x32_bf16(false, pf, false, vf,
                                                           (short)0, O[c], false, false);
        }
    }
    // ---- finalize: divide by row sum, store bf16 ----
#pragma unroll
    for (int i = 0; i < 8; ++i) {
        float rl = 1.0f / lsum[i];
        size_t trow = (size_t)(t0 + i + 8 * half);
#pragma unroll
        for (int c = 0; c < 8; ++c)
            aout[trow * HD + h * DH + c * 16 + lrow] = (__bf16)(O[c][i] * rl);
    }
}

// ============================================================
// host-side launcher
// ============================================================
extern "C" void kernel_launch(void* const* d_in, const int* in_sizes, int n_in,
                              void* d_out, int out_size, void* d_ws, size_t ws_size,
                              hipStream_t stream) {
    const float* hs   = (const float*)d_in[0];
    const float* cosb = (const float*)d_in[1];
    const float* sinb = (const float*)d_in[2];
    const float* Wq   = (const float*)d_in[3];
    const float* Wk   = (const float*)d_in[4];
    const float* Wv   = (const float*)d_in[5];
    const float* Wo   = (const float*)d_in[6];
    const float* qnw  = (const float*)d_in[7];
    const float* knw  = (const float*)d_in[8];
    float* out = (float*)d_out;

    char* w = (char*)d_ws;
    auto carve = [&](size_t bytes) { void* p = (void*)w; w += (bytes + 255) & ~(size_t)255; return p; };

    __bf16* hsb  = (__bf16*)carve((size_t)TT * HID * 2);
    __bf16* wqt  = (__bf16*)carve((size_t)HD  * HID * 2);
    __bf16* wkt  = (__bf16*)carve((size_t)KVD * HID * 2);
    __bf16* wvt  = (__bf16*)carve((size_t)KVD * HID * 2);
    __bf16* wot  = (__bf16*)carve((size_t)HID * HD  * 2);
    float*  qf   = (float* )carve((size_t)TT * HD  * 4);
    float*  kf   = (float* )carve((size_t)TT * KVD * 4);
    float*  vf   = (float* )carve((size_t)TT * KVD * 4);
    __bf16* qhat = (__bf16*)carve((size_t)TT * HD  * 2);
    __bf16* khat = (__bf16*)carve((size_t)TT * KVD * 2);
    __bf16* vtb  = (__bf16*)carve((size_t)TT * KVD * 2);
    __bf16* aout = (__bf16*)carve((size_t)TT * HD  * 2);

    // 1) conversions / transposes
    {
        int n = TT * HID;
        k_f32_to_bf16<<<(n + 255) / 256, 256, 0, stream>>>(hs, hsb, n);
        int nq = HD * HID;
        k_transpose_bf16<<<(nq + 255) / 256, 256, 0, stream>>>(Wq, wqt, HID, HD);
        int nk = KVD * HID;
        k_transpose_bf16<<<(nk + 255) / 256, 256, 0, stream>>>(Wk, wkt, HID, KVD);
        k_transpose_bf16<<<(nk + 255) / 256, 256, 0, stream>>>(Wv, wvt, HID, KVD);
        int no = HID * HD;
        k_transpose_bf16<<<(no + 255) / 256, 256, 0, stream>>>(Wo, wot, HD, HID);
    }
    // 2) QKV projections (32x64 macro tiles per wave)
    {
        int tiles_q = (TT / 32) * (HD / 64);
        k_gemm_wmma<<<(tiles_q + 7) / 8, 256, 0, stream>>>(hsb, wqt, qf, TT, HD, HID);
        int tiles_k = (TT / 32) * (KVD / 64);
        k_gemm_wmma<<<(tiles_k + 7) / 8, 256, 0, stream>>>(hsb, wkt, kf, TT, KVD, HID);
        k_gemm_wmma<<<(tiles_k + 7) / 8, 256, 0, stream>>>(hsb, wvt, vf, TT, KVD, HID);
    }
    // 3) norm + rope + V transpose
    {
        long rq = (long)TT * NH;
        k_rmsnorm_rope<<<(int)((rq + 7) / 8), 256, 0, stream>>>(qf, qnw, cosb, sinb, qhat, NH);
        long rk = (long)TT * NKV;
        k_rmsnorm_rope<<<(int)((rk + 7) / 8), 256, 0, stream>>>(kf, knw, cosb, sinb, khat, NKV);
        int nv = TT * KVD;
        k_v_transpose<<<(nv + 255) / 256, 256, 0, stream>>>(vf, vtb, nv);
    }
    // 4) attention (async-LDS double-buffered)
    {
        dim3 grid(SS / 16, BB * NH);
        k_flash_attn<<<grid, 32, 0, stream>>>(qhat, khat, vtb, aout);
    }
    // 5) output projection
    {
        int tiles_o = (TT / 32) * (HID / 64);
        k_gemm_wmma<<<(tiles_o + 7) / 8, 256, 0, stream>>>(aout, wot, out, TT, HID, HD);
    }
    (void)in_sizes; (void)n_in; (void)out_size; (void)ws_size;
}
